// GINetReconEmbeddingZeros_4183298146471
// MI455X (gfx1250) — matrix operand order repaired and checked by reading, not compiled, not run on hardware
//
#include <hip/hip_runtime.h>
#include <math.h>

#define N_NODES 50000
#define N_EDGES 200000
#define N_GRAPHS 2000
#define EMB 300
#define EMB2 600
#define FEATD 512
#define HID 256
#define NUM_LAYER 5
#define NUM_BOND_TYPE 5
#define NUM_BOND_DIR 3
#define EPSV 1e-5f

typedef __attribute__((ext_vector_type(16))) __bf16 v16bf;
typedef __attribute__((ext_vector_type(8)))  float  v8f;

__device__ __forceinline__ void atomAddF(float* p, float v) {
  __hip_atomic_fetch_add(p, v, __ATOMIC_RELAXED, __HIP_MEMORY_SCOPE_AGENT);
}

// ----------------------------- elementwise -----------------------------

__global__ void fill_zero_k(float* p, long n) {
  long i = (long)blockIdx.x * blockDim.x + threadIdx.x;
  if (i < n) p[i] = 0.f;
}

__global__ void init_embed_k(float* __restrict__ h, const int* __restrict__ x,
                             const float* __restrict__ e1, const float* __restrict__ e2) {
  long i = (long)blockIdx.x * blockDim.x + threadIdx.x;
  if (i >= (long)N_NODES * EMB) return;
  int node = (int)(i / EMB), j = (int)(i % EMB);
  h[i] = e1[(long)x[node * 2 + 0] * EMB + j] + e2[(long)x[node * 2 + 1] * EMB + j];
}

// agg = h + self_loop_edge_embedding
__global__ void agg_init_k(float* __restrict__ agg, const float* __restrict__ h,
                           const float* __restrict__ se1, const float* __restrict__ se2) {
  long i = (long)blockIdx.x * blockDim.x + threadIdx.x;
  if (i >= (long)N_NODES * EMB) return;
  int j = (int)(i % EMB);
  agg[i] = h[i] + se1[j] + se2[j];
}

// one block per edge: agg[dst] += h[src] + edge_embedding
__global__ void edge_msg_k(float* __restrict__ agg, const float* __restrict__ h,
                           const int* __restrict__ src, const int* __restrict__ dst,
                           const int* __restrict__ ea,
                           const float* __restrict__ ee1, const float* __restrict__ ee2) {
  int e = blockIdx.x;
  int s = src[e], d = dst[e];
  const float* hs = h + (long)s * EMB;
  const float* e1 = ee1 + (long)ea[e * 2 + 0] * EMB;
  const float* e2 = ee2 + (long)ea[e * 2 + 1] * EMB;
  float* ag = agg + (long)d * EMB;
  for (int j = threadIdx.x; j < EMB; j += blockDim.x)
    atomAddF(&ag[j], hs[j] + e1[j] + e2[j]);
}

// ----------------------------- WMMA GEMM -----------------------------
// C[M,Nc] = act(A[M,K] @ W[K,Nc] + bias)   act: 0=none 1=relu 2=softplus
// block: 256 thr = 8 waves -> 64x128 tile; each wave: one A frag, four B frags,
// four v_wmma_f32_16x16x32_bf16 per K-step. fp32 -> bf16 staged through LDS.

__global__ __launch_bounds__(256)
void wmma_gemm_k(const float* __restrict__ A, const float* __restrict__ W,
                 const float* __restrict__ bias, float* __restrict__ C,
                 int M, int K, int Nc, int act) {
  __shared__ __bf16 sA[64 * 32];    // [row][k]
  __shared__ __bf16 sB[128 * 32];   // [n][k]  (transposed: K-pairs contiguous)

  const int tid  = threadIdx.x;
  const int lane = tid & 31;
  const int wave = tid >> 5;
  const int ll   = lane & 15;
  const int hi   = lane >> 4;
  const int tM   = wave & 3;   // 0..3 (M sub-tile of 16)
  const int tN4  = wave >> 2;  // 0..1 (group of four N sub-tiles)
  const int mBase = blockIdx.x * 64;
  const int nBase = blockIdx.y * 128;

  v8f acc[4] = {{}, {}, {}, {}};

  for (int k0 = 0; k0 < K; k0 += 32) {
    const bool full = (mBase + 64 <= M) && (nBase + 128 <= Nc) && (k0 + 32 <= K);
    if (full) {
      // fast path: unconditional vectorized staging (no EXEC divergence)
#pragma unroll
      for (int it = 0; it < 2; ++it) {
        int q = tid + it * 256;            // 0..511 float4s of A tile (64x32)
        int r = q >> 3, kq = (q & 7) << 2;
        float4 f = *(const float4*)(A + (long)(mBase + r) * K + (k0 + kq));
        __bf16* dp = &sA[r * 32 + kq];
        dp[0] = (__bf16)f.x; dp[1] = (__bf16)f.y;
        dp[2] = (__bf16)f.z; dp[3] = (__bf16)f.w;
      }
#pragma unroll
      for (int it = 0; it < 4; ++it) {
        int q = tid + it * 256;            // 0..1023 float4s of W tile (32x128)
        int kk = q >> 5, nq = (q & 31) << 2;
        float4 f = *(const float4*)(W + (long)(k0 + kk) * Nc + (nBase + nq));
        sB[(nq + 0) * 32 + kk] = (__bf16)f.x;
        sB[(nq + 1) * 32 + kk] = (__bf16)f.y;
        sB[(nq + 2) * 32 + kk] = (__bf16)f.z;
        sB[(nq + 3) * 32 + kk] = (__bf16)f.w;
      }
      if (k0 + 32 < K) {  // prefetch next K-slab (global_prefetch_b8)
        int pr = mBase + (tid & 63);
        int pk = k0 + 32 + (tid & 31);
        if (pk >= K) pk = K - 1;
        __builtin_prefetch(A + (long)pr * K + pk, 0, 1);
        __builtin_prefetch(W + (long)pk * Nc + nBase + ((2 * tid) & 127), 0, 1);
      }
    } else {
      // slow path: clamped indices + select; keep code small (no full unroll)
#pragma unroll 1
      for (int it = 0; it < 8; ++it) {
        int q = tid + it * 256;            // A tile elements (64x32)
        int r = q >> 5, kk = q & 31;
        int gm = mBase + r, gk = k0 + kk;
        int gmc = gm < M ? gm : M - 1;
        int gkc = gk < K ? gk : K - 1;
        float v = A[(long)gmc * K + gkc];
        sA[q] = (__bf16)((gm < M && gk < K) ? v : 0.f);
      }
#pragma unroll 1
      for (int it = 0; it < 16; ++it) {
        int q = tid + it * 256;            // W tile elements (32x128)
        int kk = q >> 7, nn = q & 127;
        int gk = k0 + kk, gn = nBase + nn;
        int gkc = gk < K ? gk : K - 1;
        int gnc = gn < Nc ? gn : Nc - 1;
        float v = W[(long)gkc * Nc + gnc];
        sB[nn * 32 + kk] = (__bf16)((gk < K && gn < Nc) ? v : 0.f);
      }
    }
    __syncthreads();

    union Frag { v16bf v; unsigned u[8]; } fa, fb;
    const int ar = tM * 16 + ll;           // A row within block tile
#pragma unroll
    for (int r = 0; r < 8; ++r) {
      // A 16x32 bf16 striping: lanes0-15 V0-3:K0..7 V4-7:K16..23; lanes16-31 +8
      int ka = ((r < 4) ? 0 : 16) + hi * 8 + 2 * (r & 3);
      fa.u[r] = *(const unsigned*)(&sA[ar * 32 + ka]);
    }
#pragma unroll
    for (int sub = 0; sub < 4; ++sub) {
      const int bc = tN4 * 64 + sub * 16 + ll;   // B column within block tile
#pragma unroll
      for (int r = 0; r < 8; ++r) {
        // B 32x16 bf16 striping: lanes0-15 V0-3:K0..7 V4-7:K16..23; lanes16-31 +8
        int kb = ((r < 4) ? 0 : 16) + hi * 8 + 2 * (r & 3);
        fb.u[r] = *(const unsigned*)(&sB[bc * 32 + kb]);
      }
      acc[sub] = __builtin_amdgcn_wmma_f32_16x16x32_bf16(false, fa.v, false, fb.v,
                                                         (short)0, acc[sub], false, false);
    }
    __syncthreads();
  }

#pragma unroll
  for (int sub = 0; sub < 4; ++sub) {
    const int col = nBase + tN4 * 64 + sub * 16 + ll;
    if (col < Nc) {
      float bv = bias[col];
#pragma unroll
      for (int r = 0; r < 8; ++r) {
        int row = mBase + tM * 16 + hi * 8 + r;
        if (row < M) {
          float v = acc[sub][r] + bv;
          if (act == 1)      v = fmaxf(v, 0.f);
          else if (act == 2) v = fmaxf(v, 0.f) + log1pf(expf(-fabsf(v)));
          C[(long)row * Nc + col] = v;
        }
      }
    }
  }
}

// ----------------------------- BatchNorm -----------------------------

__global__ void bn_stats_k(const float* __restrict__ h, float* __restrict__ mean,
                           float* __restrict__ rstd) {
  int f = blockIdx.x;  // feature
  float s = 0.f, ss = 0.f;
  for (int i = threadIdx.x; i < N_NODES; i += blockDim.x) {
    float v = h[(long)i * EMB + f];
    s += v; ss += v * v;
  }
  __shared__ float rs[256], rq[256];
  rs[threadIdx.x] = s; rq[threadIdx.x] = ss;
  __syncthreads();
  for (int off = 128; off > 0; off >>= 1) {
    if ((int)threadIdx.x < off) {
      rs[threadIdx.x] += rs[threadIdx.x + off];
      rq[threadIdx.x] += rq[threadIdx.x + off];
    }
    __syncthreads();
  }
  if (threadIdx.x == 0) {
    float m   = rs[0] / (float)N_NODES;
    float var = rq[0] / (float)N_NODES - m * m;
    mean[f] = m;
    rstd[f] = rsqrtf(var + EPSV);
  }
}

__global__ void bn_apply_k(float* __restrict__ out, const float* __restrict__ hin,
                           const float* __restrict__ mean, const float* __restrict__ rstd,
                           const float* __restrict__ gamma, const float* __restrict__ beta,
                           int relu) {
  long i = (long)blockIdx.x * blockDim.x + threadIdx.x;
  if (i >= (long)N_NODES * EMB) return;
  int j = (int)(i % EMB);
  float v = (hin[i] - mean[j]) * rstd[j] * gamma[j] + beta[j];
  if (relu) v = fmaxf(v, 0.f);
  out[i] = v;
}

// ----------------------------- pooling / head -----------------------------

__global__ void pool_count_k(float* __restrict__ cnt, const int* __restrict__ batch) {
  long i = (long)blockIdx.x * blockDim.x + threadIdx.x;
  if (i >= N_NODES) return;
  atomAddF(&cnt[batch[i]], 1.f);
}

__global__ void pool_sum_k(float* __restrict__ pooled, const float* __restrict__ h,
                           const int* __restrict__ batch) {
  long i = (long)blockIdx.x * blockDim.x + threadIdx.x;
  if (i >= (long)N_NODES * EMB) return;
  int node = (int)(i / EMB), j = (int)(i % EMB);
  atomAddF(&pooled[(long)batch[node] * EMB + j], h[i]);
}

__global__ void pool_div_k(float* __restrict__ pooled, const float* __restrict__ cnt) {
  long i = (long)blockIdx.x * blockDim.x + threadIdx.x;
  if (i >= (long)N_GRAPHS * EMB) return;
  int g = (int)(i / EMB);
  pooled[i] /= fmaxf(cnt[g], 1.f);
}

__global__ void pred_k(const float* __restrict__ z, const float* __restrict__ w,
                       const float* __restrict__ b, float* __restrict__ out) {
  int g = blockIdx.x;
  __shared__ float rs[HID];
  rs[threadIdx.x] = z[(long)g * HID + threadIdx.x] * w[threadIdx.x];
  __syncthreads();
  for (int off = HID / 2; off > 0; off >>= 1) {
    if ((int)threadIdx.x < off) rs[threadIdx.x] += rs[threadIdx.x + off];
    __syncthreads();
  }
  if (threadIdx.x == 0) out[g] = rs[0] + b[0];
}

// ----------------------------- launcher -----------------------------

extern "C" void kernel_launch(void* const* d_in, const int* in_sizes, int n_in,
                              void* d_out, int out_size, void* d_ws, size_t ws_size,
                              hipStream_t stream) {
  const int*   x          = (const int*)d_in[0];
  const int*   edge_index = (const int*)d_in[1];
  const int*   edge_attr  = (const int*)d_in[2];
  const int*   batch      = (const int*)d_in[3];
  const float* x_emb1     = (const float*)d_in[5];
  const float* x_emb2     = (const float*)d_in[6];
  const float* edge_emb1  = (const float*)d_in[7];
  const float* edge_emb2  = (const float*)d_in[8];
  const float* W1    = (const float*)d_in[9];
  const float* b1    = (const float*)d_in[10];
  const float* W2    = (const float*)d_in[11];
  const float* b2    = (const float*)d_in[12];
  const float* gamma = (const float*)d_in[13];
  const float* beta  = (const float*)d_in[14];
  const float* featW = (const float*)d_in[15];
  const float* featb = (const float*)d_in[16];
  const float* pW1 = (const float*)d_in[17];
  const float* pb1 = (const float*)d_in[18];
  const float* pW2 = (const float*)d_in[19];
  const float* pb2 = (const float*)d_in[20];
  const float* pW3 = (const float*)d_in[21];
  const float* pb3 = (const float*)d_in[22];

  float* ws     = (float*)d_ws;
  float* h      = ws;                               // N*EMB
  float* agg    = h + (long)N_NODES * EMB;          // N*EMB (also GEMM2 output)
  float* t      = agg + (long)N_NODES * EMB;        // N*EMB2
  float* mean   = t + (long)N_NODES * EMB2;         // EMB
  float* rstd   = mean + EMB;                       // EMB
  float* cnt    = rstd + EMB;                       // G
  float* pooled = cnt + N_GRAPHS;                   // G*EMB (contiguous with cnt)
  float* feat   = pooled + (long)N_GRAPHS * EMB;    // G*FEATD
  float* z1     = feat + (long)N_GRAPHS * FEATD;    // G*HID
  float* z2     = z1 + (long)N_GRAPHS * HID;        // G*HID

  float* h_node = (float*)d_out;                    // N*EMB
  float* pred   = h_node + (long)N_NODES * EMB;     // G

  const long NE = (long)N_NODES * EMB;
  const int TB = 256;
  auto blks = [](long n) { return (unsigned)((n + 255) / 256); };

  init_embed_k<<<blks(NE), TB, 0, stream>>>(h, x, x_emb1, x_emb2);

  for (int l = 0; l < NUM_LAYER; ++l) {
    const float* ee1 = edge_emb1 + (long)l * NUM_BOND_TYPE * EMB;
    const float* ee2 = edge_emb2 + (long)l * NUM_BOND_DIR * EMB;
    agg_init_k<<<blks(NE), TB, 0, stream>>>(agg, h, ee1 + 4 * EMB, ee2);
    edge_msg_k<<<N_EDGES, TB, 0, stream>>>(
        agg, h, edge_index, edge_index + N_EDGES, edge_attr, ee1, ee2);

    dim3 g1((N_NODES + 63) / 64, (EMB2 + 127) / 128);
    wmma_gemm_k<<<g1, TB, 0, stream>>>(agg, W1 + (long)l * EMB * EMB2,
                                       b1 + (long)l * EMB2, t, N_NODES, EMB, EMB2, 1);
    dim3 g2((N_NODES + 63) / 64, (EMB + 127) / 128);
    wmma_gemm_k<<<g2, TB, 0, stream>>>(t, W2 + (long)l * EMB2 * EMB,
                                       b2 + (long)l * EMB, agg, N_NODES, EMB2, EMB, 0);

    bn_stats_k<<<EMB, TB, 0, stream>>>(agg, mean, rstd);
    float* dst = (l == NUM_LAYER - 1) ? h_node : h;
    bn_apply_k<<<blks(NE), TB, 0, stream>>>(dst, agg, mean, rstd,
                                            gamma + (long)l * EMB, beta + (long)l * EMB,
                                            (l < NUM_LAYER - 1) ? 1 : 0);
  }

  // zero cnt + pooled (contiguous)
  fill_zero_k<<<blks((long)N_GRAPHS * (EMB + 1)), TB, 0, stream>>>(cnt, (long)N_GRAPHS * (EMB + 1));
  pool_count_k<<<blks(N_NODES), TB, 0, stream>>>(cnt, batch);
  pool_sum_k<<<blks(NE), TB, 0, stream>>>(pooled, h_node, batch);
  pool_div_k<<<blks((long)N_GRAPHS * EMB), TB, 0, stream>>>(pooled, cnt);

  dim3 gf((N_GRAPHS + 63) / 64, (FEATD + 127) / 128);
  wmma_gemm_k<<<gf, TB, 0, stream>>>(pooled, featW, featb, feat, N_GRAPHS, EMB, FEATD, 0);
  dim3 gz((N_GRAPHS + 63) / 64, (HID + 127) / 128);
  wmma_gemm_k<<<gz, TB, 0, stream>>>(feat, pW1, pb1, z1, N_GRAPHS, FEATD, HID, 2);
  wmma_gemm_k<<<gz, TB, 0, stream>>>(z1, pW2, pb2, z2, N_GRAPHS, HID, HID, 2);
  pred_k<<<N_GRAPHS, HID, 0, stream>>>(z2, pW3, pb3, pred);
}
